// SAModule_45938970198691
// MI455X (gfx1250) — matrix operand order repaired and checked by reading, not compiled, not run on hardware
//
#include <hip/hip_runtime.h>

typedef __bf16 bf16;
typedef __attribute__((ext_vector_type(16))) bf16 v16bf;
typedef __attribute__((ext_vector_type(8)))  bf16 v8bf;
typedef __attribute__((ext_vector_type(8)))  float v8f;

#define B_   8
#define N_   8192
#define S_   2048
#define K_   32
#define CF   64
#define KP1  96      // 67 padded to 96 (3 x K=32 wmma steps)
#define C1   128
#define C2   256
#define M_   (B_*S_*K_)   // 524288 rows
#define EPSBN 1e-5f

// ---------------------------------------------------------------------------
// A-fragment (16x32 bf16, MxK): lane<16 holds K {0..15}, lane>=16 holds
// K {16..31}, for row M = lane&15 (same k order as the B fragment).
// ---------------------------------------------------------------------------
__device__ __forceinline__ v16bf load_a_frag(const bf16* p /*row base + k off*/) {
    v8bf lo = *(const v8bf*)(p);
    v8bf hi = *(const v8bf*)(p + 8);
    v16bf a;
#pragma unroll
    for (int i = 0; i < 8; ++i) { a[i] = lo[i]; a[i + 8] = hi[i]; }
    return a;
}

// ---------------------------------------------------------------------------
// Kernel 1: serial farthest-point sampling, one block per batch.
// 512 threads x 16 points each, point cache + running min-dist in registers.
// ---------------------------------------------------------------------------
__global__ __launch_bounds__(512) void fps_kernel(const float* __restrict__ xyz,
                                                  float* __restrict__ out_xyz) {
    const int b = blockIdx.x;
    const int t = threadIdx.x;
    const float* xb = xyz + (size_t)b * N_ * 3;
    float px[16], py[16], pz[16], dist[16];
#pragma unroll
    for (int i = 0; i < 16; ++i) {
        int n = t * 16 + i;
        px[i] = xb[n * 3 + 0];
        py[i] = xb[n * 3 + 1];
        pz[i] = xb[n * 3 + 2];
        dist[i] = 3.402823466e38f;
    }
    __shared__ float lastc[3];
    __shared__ float wv[16];
    __shared__ int   wi[16];
    __shared__ int   winner;
    int last = 0;
    for (int it = 0; it < S_; ++it) {
        if (t == (last >> 4)) {            // owner of current sample
            int sl = last & 15;
            float* o = out_xyz + ((size_t)b * S_ + it) * 3;
            o[0] = px[sl]; o[1] = py[sl]; o[2] = pz[sl];
            lastc[0] = px[sl]; lastc[1] = py[sl]; lastc[2] = pz[sl];
        }
        __syncthreads();
        const float lx = lastc[0], ly = lastc[1], lz = lastc[2];
        float best = -1.0f; int bi = 0;
#pragma unroll
        for (int i = 0; i < 16; ++i) {
            float dx = px[i] - lx, dy = py[i] - ly, dz = pz[i] - lz;
            float d  = dx * dx + dy * dy + dz * dz;
            float dm = fminf(dist[i], d);
            dist[i] = dm;
            if (dm > best) { best = dm; bi = t * 16 + i; }
        }
#pragma unroll
        for (int off = 16; off > 0; off >>= 1) {   // wave32 argmax
            float ov = __shfl_down(best, off);
            int   oi = __shfl_down(bi,   off);
            if (ov > best) { best = ov; bi = oi; }
        }
        if ((t & 31) == 0) { wv[t >> 5] = best; wi[t >> 5] = bi; }
        __syncthreads();
        if (t < 32) {                               // wave 0: reduce 16 waves
            float v2 = (t < 16) ? wv[t] : -2.0f;
            int   i2 = (t < 16) ? wi[t] : 0;
#pragma unroll
            for (int off = 16; off > 0; off >>= 1) {
                float ov = __shfl_down(v2, off);
                int   oi = __shfl_down(i2, off);
                if (ov > v2) { v2 = ov; i2 = oi; }
            }
            if (t == 0) winner = i2;
        }
        __syncthreads();
        last = winner;
    }
}

// ---------------------------------------------------------------------------
// Kernel 2: brute-force 32-NN + gather + concat, one block (256 thr) per query.
// Distances in LDS (32KB); 32 masked argmin passes; writes bf16 row of 96
// (3 xyz-diff | 64 feat | 29 zero pad) per neighbor.
// ---------------------------------------------------------------------------
__global__ __launch_bounds__(256) void knn_group_kernel(const float* __restrict__ xyz,
                                                        const float* __restrict__ feat,
                                                        const float* __restrict__ nxyz,
                                                        bf16* __restrict__ xg) {
    const int q = blockIdx.x;          // b*S + s
    const int b = q >> 11;
    const int t = threadIdx.x;
    __shared__ float d2[N_];
    __shared__ float rv[256];
    __shared__ int   ri[256];
    __shared__ int   nbr[K_];
    const float qx = nxyz[q * 3 + 0], qy = nxyz[q * 3 + 1], qz = nxyz[q * 3 + 2];
    const float* xb = xyz + (size_t)b * N_ * 3;
    for (int n = t; n < N_; n += 256) {
        float dx = xb[n * 3 + 0] - qx;
        float dy = xb[n * 3 + 1] - qy;
        float dz = xb[n * 3 + 2] - qz;
        d2[n] = dx * dx + dy * dy + dz * dz;
    }
    __syncthreads();
    for (int k = 0; k < K_; ++k) {
        float best = 3.4e38f; int bi = t;
        for (int n = t; n < N_; n += 256) {
            float v = d2[n];
            if (v < best) { best = v; bi = n; }
        }
        rv[t] = best; ri[t] = bi;
        __syncthreads();
        for (int off = 128; off > 0; off >>= 1) {
            if (t < off) {
                if (rv[t + off] < rv[t]) { rv[t] = rv[t + off]; ri[t] = ri[t + off]; }
            }
            __syncthreads();
        }
        if (t == 0) { nbr[k] = ri[0]; d2[ri[0]] = 3.402823466e38f; }
        __syncthreads();
    }
    const float* fb = feat + (size_t)b * CF * N_;
    for (int e = t; e < K_ * KP1; e += 256) {
        int k = e / KP1, c = e - k * KP1;
        int n = nbr[k];
        float v;
        if (c < 3)            v = xb[n * 3 + c] - nxyz[q * 3 + c];
        else if (c < 3 + CF)  v = fb[(size_t)(c - 3) * N_ + n];
        else                  v = 0.0f;
        xg[((size_t)q * K_ + k) * KP1 + c] = (bf16)v;
    }
}

// ---------------------------------------------------------------------------
// Kernel 3: GEMM1 raw: y1[M,128] = x[M,96] @ w1p[96,128]  (bf16 in, fp32 acc).
// Block = 128 rows x 128 cols (w1 staging amortized 4x vs 32-row tiles);
// 8 waves, each wave owns one 16-col tile and all 8 16-row tiles;
// 3 K-steps x 8 m-tiles of v_wmma_f32_16x16x32_bf16 (24 per wave).
// Channel sum/sumsq accumulated via fp32 global atomics for BN1 stats.
// ---------------------------------------------------------------------------
__global__ __launch_bounds__(256) void gemm1_kernel(const bf16* __restrict__ xg,
                                                    const float* __restrict__ w1,
                                                    bf16* __restrict__ y1,
                                                    float* __restrict__ stats1) {
    __shared__ __align__(32) bf16 smA[128 * KP1];   // 24 KB
    __shared__ __align__(32) bf16 smB[C1 * KP1];    // 24 KB
    const int t = threadIdx.x, wave = t >> 5, lane = t & 31;
    const int col = lane & 15, half = lane >> 4;
    const size_t m0 = (size_t)blockIdx.x * 128;
    {   // A tile: straight 16B-vector copy (already bf16, contiguous)
        const uint4* src = (const uint4*)(xg + m0 * KP1);
        uint4* dst = (uint4*)smA;
        for (int i = t; i < (128 * KP1 * 2) / 16; i += 256) dst[i] = src[i];
    }
    for (int i = t; i < C1 * KP1; i += 256) {   // B = w1 padded 67->96, bf16
        int ch = i / KP1, c = i - ch * KP1;
        smB[i] = (c < 67) ? (bf16)w1[ch * 67 + c] : (bf16)0.0f;
    }
    __syncthreads();
    v8f acc[8];
#pragma unroll
    for (int mt = 0; mt < 8; ++mt) acc[mt] = (v8f){};
    const int ch = wave * 16 + col;
#pragma unroll
    for (int ks = 0; ks < 3; ++ks) {
        const int k0 = ks * 32;
        v16bf bb = *(const v16bf*)(&smB[ch * KP1 + k0 + half * 16]);
#pragma unroll
        for (int mt = 0; mt < 8; ++mt) {
            v16bf a = load_a_frag(&smA[(mt * 16 + col) * KP1 + k0 + half * 16]);
            acc[mt] = __builtin_amdgcn_wmma_f32_16x16x32_bf16(false, a, false, bb, (short)0, acc[mt], false, false);
        }
    }
    float s = 0.0f, ss = 0.0f;
#pragma unroll
    for (int mt = 0; mt < 8; ++mt) {
#pragma unroll
        for (int r = 0; r < 8; ++r) {
            float v = acc[mt][r];
            y1[(m0 + mt * 16 + half * 8 + r) * C1 + ch] = (bf16)v;
            s += v; ss += v * v;
        }
    }
    s  += __shfl_xor(s, 16);
    ss += __shfl_xor(ss, 16);
    if (lane < 16) {
        atomicAdd(&stats1[ch], s);
        atomicAdd(&stats1[C1 + ch], ss);
    }
}

// ---------------------------------------------------------------------------
// Kernel 4/6: finalize BN affine params: scale = g*rsqrt(var+eps),
// shift = b - mean*scale  (count = M_ rows).
// ---------------------------------------------------------------------------
__global__ void bnparams_kernel(const float* __restrict__ stats, const float* __restrict__ g,
                                const float* __restrict__ bb, float* __restrict__ bnout, int C) {
    int c = blockIdx.x * blockDim.x + threadIdx.x;
    if (c >= C) return;
    const float inv = 1.0f / (float)M_;
    float mean = stats[c] * inv;
    float var  = stats[C + c] * inv - mean * mean;
    float a    = g[c] * rsqrtf(var + EPSBN);
    bnout[c]     = a;
    bnout[C + c] = bb[c] - mean * a;
}

// ---------------------------------------------------------------------------
// Kernel 5: GEMM2: applies BN1+ReLU on the A-load (y1 raw bf16 -> normalized
// bf16), multiplies by w2[256,128]. Block = 64 rows (== TWO query points s,
// 32 neighbors each) x 128 channels (grid.y halves). Emits per-(s,ch) max/min
// over the neighbor axis (monotone BN2+ReLU+maxpool fusion) plus channel
// sum/sumsq atomics for BN2 stats. y2 is never materialized.
// ---------------------------------------------------------------------------
__global__ __launch_bounds__(256) void gemm2_kernel(const bf16* __restrict__ y1,
                                                    const float* __restrict__ w2,
                                                    const float* __restrict__ bn1,
                                                    float* __restrict__ stats2,
                                                    float* __restrict__ maxv,
                                                    float* __restrict__ minv) {
    __shared__ __align__(32) bf16 smA[64 * C1];     // 16 KB
    __shared__ __align__(32) bf16 smB[128 * C1];    // 32 KB
    const int t = threadIdx.x, wave = t >> 5, lane = t & 31;
    const int col = lane & 15, half = lane >> 4;
    const int qb = blockIdx.x;                 // pair of query points
    const size_t m0 = (size_t)qb * 64;
    const int chbase = blockIdx.y * 128;
    const float* sc1 = bn1;
    const float* sh1 = bn1 + C1;
    for (int i = t; i < 64 * C1; i += 256) {   // A: BN1 + ReLU fused on load
        int c = i & (C1 - 1);
        float v = (float)y1[m0 * C1 + i];
        v = v * sc1[c] + sh1[c];
        smA[i] = (bf16)fmaxf(v, 0.0f);
    }
    for (int i = t; i < 128 * C1; i += 256) {  // B: half of w2, bf16
        smB[i] = (bf16)w2[(size_t)chbase * C1 + i];
    }
    __syncthreads();
    v8f acc[4];
#pragma unroll
    for (int mt = 0; mt < 4; ++mt) acc[mt] = (v8f){};
    const int chl = wave * 16 + col;
#pragma unroll
    for (int ks = 0; ks < 4; ++ks) {
        const int k0 = ks * 32;
        v16bf bb = *(const v16bf*)(&smB[chl * C1 + k0 + half * 16]);
#pragma unroll
        for (int mt = 0; mt < 4; ++mt) {
            v16bf a = load_a_frag(&smA[(mt * 16 + col) * C1 + k0 + half * 16]);
            acc[mt] = __builtin_amdgcn_wmma_f32_16x16x32_bf16(false, a, false, bb, (short)0, acc[mt], false, false);
        }
    }
    const int ch = chbase + chl;
    float s = 0.0f, ss = 0.0f;
#pragma unroll
    for (int sh = 0; sh < 2; ++sh) {           // m-tiles (2*sh, 2*sh+1) = one s
        float mx = -3.4e38f, mn = 3.4e38f;
#pragma unroll
        for (int j = 0; j < 2; ++j) {
            const int mt = 2 * sh + j;
#pragma unroll
            for (int r = 0; r < 8; ++r) {
                float v = acc[mt][r];
                s += v; ss += v * v;
                mx = fmaxf(mx, v); mn = fminf(mn, v);
            }
        }
        mx = fmaxf(mx, __shfl_xor(mx, 16));    // lanes (l, l+16) = same channel,
        mn = fminf(mn, __shfl_xor(mn, 16));    // disjoint row halves -> 32-row reduce
        if (lane < 16) {
            size_t q = (size_t)qb * 2 + sh;
            maxv[q * C2 + ch] = mx;
            minv[q * C2 + ch] = mn;
        }
    }
    s  += __shfl_xor(s, 16);
    ss += __shfl_xor(ss, 16);
    if (lane < 16) {
        atomicAdd(&stats2[ch], s);
        atomicAdd(&stats2[C2 + ch], ss);
    }
}

// ---------------------------------------------------------------------------
// Kernel 7: fused BN2 + ReLU + neighbor-maxpool epilogue -> [B, 256, S].
// max_k relu(a*x+b) == relu(a*(a>=0 ? max_k x : min_k x) + b).
// ---------------------------------------------------------------------------
__global__ __launch_bounds__(256) void out_kernel(const float* __restrict__ maxv,
                                                  const float* __restrict__ minv,
                                                  const float* __restrict__ bn2,
                                                  float* __restrict__ outf) {
    size_t e = (size_t)blockIdx.x * 256 + threadIdx.x;   // [B,C2,S] flat
    int s  = (int)(e & (S_ - 1));
    int ch = (int)((e >> 11) & (C2 - 1));
    int b  = (int)(e >> 19);
    size_t q = (size_t)b * S_ + s;
    float a  = bn2[ch], sh = bn2[C2 + ch];
    float m  = (a >= 0.0f) ? maxv[q * C2 + ch] : minv[q * C2 + ch];
    outf[e] = fmaxf(m * a + sh, 0.0f);
}

// ---------------------------------------------------------------------------
extern "C" void kernel_launch(void* const* d_in, const int* in_sizes, int n_in,
                              void* d_out, int out_size, void* d_ws, size_t ws_size,
                              hipStream_t stream) {
    const float* xyz  = (const float*)d_in[0];
    const float* feat = (const float*)d_in[1];
    const float* w1   = (const float*)d_in[2];
    const float* g1   = (const float*)d_in[3];
    const float* b1   = (const float*)d_in[4];
    const float* w2   = (const float*)d_in[5];
    const float* g2   = (const float*)d_in[6];
    const float* b2   = (const float*)d_in[7];
    float* out      = (float*)d_out;
    float* out_xyz  = out;                              // [8,2048,3]
    float* out_feat = out + (size_t)B_ * S_ * 3;        // [8,256,2048]

    char* ws = (char*)d_ws;
    size_t o = 0;
    bf16*  xg     = (bf16*)(ws + o);  o += (size_t)M_ * KP1 * 2;     //  96 MB
    bf16*  y1     = (bf16*)(ws + o);  o += (size_t)M_ * C1 * 2;      // 128 MB
    float* stats1 = (float*)(ws + o); o += 2 * C1 * 4;
    float* bn1    = (float*)(ws + o); o += 2 * C1 * 4;
    float* stats2 = (float*)(ws + o); o += 2 * C2 * 4;
    float* bn2    = (float*)(ws + o); o += 2 * C2 * 4;
    float* maxv   = (float*)(ws + o); o += (size_t)B_ * S_ * C2 * 4; //  16 MB
    float* minv   = (float*)(ws + o); o += (size_t)B_ * S_ * C2 * 4; //  16 MB

    hipMemsetAsync(stats1, 0, 2 * C1 * 4, stream);
    hipMemsetAsync(stats2, 0, 2 * C2 * 4, stream);

    fps_kernel      <<<B_, 512, 0, stream>>>(xyz, out_xyz);
    knn_group_kernel<<<B_ * S_, 256, 0, stream>>>(xyz, feat, out_xyz, xg);
    gemm1_kernel    <<<M_ / 128, 256, 0, stream>>>(xg, w1, y1, stats1);
    bnparams_kernel <<<1, 128, 0, stream>>>(stats1, g1, b1, bn1, C1);
    gemm2_kernel    <<<dim3(B_ * S_ / 2, 2), 256, 0, stream>>>(y1, w2, bn1, stats2, maxv, minv);
    bnparams_kernel <<<1, 256, 0, stream>>>(stats2, g2, b2, bn2, C2);
    out_kernel      <<<(B_ * C2 * S_) / 256, 256, 0, stream>>>(maxv, minv, bn2, out_feat);
}
